// Dynamic_MLP_C_12008728560063
// MI455X (gfx1250) — matrix operand order repaired and checked by reading, not compiled, not run on hardware
//
#include <hip/hip_runtime.h>
#include <hip/hip_bf16.h>

// MI455X / gfx1250, wave32. All GEMMs via v_wmma_f32_16x16x32_bf16 (f32 accum).
// Compute-bound fused kernel: ~19 GFLOP total, ~25 MB HBM traffic.
// W22 tile staging uses GLOBAL_LOAD_ASYNC_TO_LDS_B128 (ASYNCcnt) to overlap
// the 8KB/iteration LDS fill with WMMA compute.

typedef __bf16 bf16_t;
typedef __attribute__((ext_vector_type(8)))  __bf16 v8bf;
typedef __attribute__((ext_vector_type(16))) __bf16 v16bf;
typedef __attribute__((ext_vector_type(8)))  float  v8f;

__device__ __forceinline__ v8f wmma_bf16(v16bf a, v16bf b, v8f c) {
  // 8 args: (neg_a, A, neg_b, B, c_mod, C, reuse_a, reuse_b)
  return __builtin_amdgcn_wmma_f32_16x16x32_bf16(false, a, false, b, (short)0, c,
                                                 false, false);
}

// Async 16B global->LDS copy, tracked by ASYNCcnt (no VGPR data path).
// lds_off: wave-relative LDS byte address = low 32 bits of generic shared ptr.
__device__ __forceinline__ void async_g2l_b128(unsigned int lds_off, const void* g) {
  asm volatile("global_load_async_to_lds_b128 %0, %1, off"
               :: "v"(lds_off), "v"((unsigned long long)(uintptr_t)g)
               : "memory");
}
__device__ __forceinline__ void wait_async0() {
  asm volatile("s_wait_asynccnt 0x0" ::: "memory");
}

// A fragment (16x32 bf16), row-major activations in LDS, row stride ldk (bf16).
// lane m = lane&15 selects row M; kh = lane>>4 selects K-half.
// lanes 0-15: K = k0+{0..15}; lanes 16-31: K = k0+{16..31}.
__device__ __forceinline__ v16bf load_a_frag(const bf16_t* A, int ldk, int k0,
                                             int m, int kh) {
  const bf16_t* p = A + m * ldk + k0 + kh * 16;
  v8bf lo = *(const v8bf*)(p);
  v8bf hi = *(const v8bf*)(p + 8);
  v16bf a;
#pragma unroll
  for (int e = 0; e < 8; ++e) { a[e] = lo[e]; a[e + 8] = hi[e]; }
  return a;
}

// B fragment (32x16 bf16) from LDS weight stored TRANSPOSED: BT[n][k] (k contig).
// lane holds column n = n0 + lane&15, K = k0 + kh*16 + {0..15} (32B contiguous).
__device__ __forceinline__ v16bf load_b_frag(const bf16_t* BT, int ldk, int n,
                                             int k0, int kh) {
  return *(const v16bf*)(BT + n * ldk + k0 + kh * 16);
}

// One-time: W22[64,4096] f32 -> W22T[4096,64] bf16 (K-contiguous rows) in ws.
__global__ __launch_bounds__(256) void w22_transpose_kernel(const float* __restrict__ W22,
                                                            bf16_t* __restrict__ W22T) {
  int idx = blockIdx.x * 256 + threadIdx.x;   // 4096*64 elements
  int c = idx >> 6;          // output row  (0..4095)
  int k = idx & 63;          // output col  (0..63)
  W22T[idx] = (bf16_t)W22[k * 4096 + c];
}

#define TOK_PER_WG 128
#define SMEM_BYTES 163840

__global__ __launch_bounds__(256) void dyn_mlp_fused_kernel(
    const float* __restrict__ img,  const float* __restrict__ loc,
    const float* __restrict__ W11,  const float* __restrict__ b11,
    const float* __restrict__ W12,  const float* __restrict__ b12,
    const float* __restrict__ W21,  const float* __restrict__ b21,
    const float* __restrict__ b22,
    const float* __restrict__ lng,  const float* __restrict__ lnb,
    const float* __restrict__ W3,   const float* __restrict__ b3,
    const bf16_t* __restrict__ W22T, float* __restrict__ out) {
  extern __shared__ char smem[];
  bf16_t* sW11T  = (bf16_t*)(smem);            // [64][128] bf16, 16KB
  bf16_t* sW21T  = (bf16_t*)(smem + 16384);    // [64][128]
  bf16_t* sW12T  = (bf16_t*)(smem + 32768);    // [64][64],  8KB
  bf16_t* sW3T   = (bf16_t*)(smem + 40960);    // [64][64]
  bf16_t* sStage = (bf16_t*)(smem + 49152);    // [2][64][64] double buffer, 16KB

  const int tid  = threadIdx.x;
  const int wave = tid >> 5;
  const int lane = tid & 31;
  const int m  = lane & 15;    // row within 16 (A) / col within 16 (B,C)
  const int kh = lane >> 4;    // K-half selector

  char* myBase = smem + 65536 + wave * 12288;  // per-wave scratch
  bf16_t* sCat  = (bf16_t*)(myBase);           // [16][128] bf16 (pool +0)
  bf16_t* sW11o = (bf16_t*)(myBase + 4096);    // [16][64]
  bf16_t* sW21o = (bf16_t*)(myBase + 6144);    // [16][64]
  bf16_t* sXr   = (bf16_t*)(myBase);           // [16][64], aliases dead sCat
  float*  sw12  = (float*)(myBase + 8192);     // [16][64] f32 scales

  // LDS byte offsets for async-to-LDS staging (low 32 bits of generic ptr)
  const unsigned int stage_off0 =
      (unsigned int)(uintptr_t)sStage + (unsigned int)tid * 16u;

  // ---- stage weights (transposed, bf16) into LDS ----
  for (int idx = tid; idx < 64 * 128; idx += 256) {
    int n = idx >> 7, k = idx & 127;
    sW11T[idx] = (bf16_t)W11[k * 64 + n];
    sW21T[idx] = (bf16_t)W21[k * 64 + n];
  }
  for (int idx = tid; idx < 64 * 64; idx += 256) {
    int n = idx >> 6, k = idx & 63;
    sW12T[idx] = (bf16_t)W12[k * 64 + n];
    sW3T[idx]  = (bf16_t)W3[k * 64 + n];
  }

  // ---- prologue: async-stage W22T rows [0,64) into buffer 0 ----
  {
    const char* src = (const char*)W22T + (size_t)tid * 16;
    async_g2l_b128(stage_off0,         src);
    async_g2l_b128(stage_off0 + 4096u, src + 4096);
  }

  __syncthreads();  // weights ready (async stage still in flight)

  const int tok0 = blockIdx.x * TOK_PER_WG + wave * 16;

  // ---- cat = [img | loc] for this wave's 16 tokens, bf16 row-major ----
  for (int idx = lane; idx < 16 * 128; idx += 32) {
    int t = idx >> 7, c = idx & 127;
    float v = (c < 64) ? img[(size_t)(tok0 + t) * 64 + c]
                       : loc[(size_t)(tok0 + t) * 64 + (c - 64)];
    sCat[idx] = (bf16_t)v;
  }

  // ---- w11 = relu(cat@W11+b11), w21 = relu(cat@W21+b21) ----
#pragma unroll
  for (int tile = 0; tile < 4; ++tile) {
    int n16 = tile * 16 + m;
    float bias1 = b11[n16], bias2 = b21[n16];
    v8f c1, c2;
#pragma unroll
    for (int r = 0; r < 8; ++r) { c1[r] = bias1; c2[r] = bias2; }
#pragma unroll
    for (int ks = 0; ks < 4; ++ks) {
      v16bf a   = load_a_frag(sCat, 128, ks * 32, m, kh);
      v16bf bb1 = load_b_frag(sW11T, 128, n16, ks * 32, kh);
      v16bf bb2 = load_b_frag(sW21T, 128, n16, ks * 32, kh);
      c1 = wmma_bf16(a, bb1, c1);
      c2 = wmma_bf16(a, bb2, c2);
    }
#pragma unroll
    for (int r = 0; r < 8; ++r) {
      int t = r + 8 * kh;  // C layout: lanes>=16 hold M=8..15
      sW11o[t * 64 + n16] = (bf16_t)fmaxf(c1[r], 0.0f);
      sW21o[t * 64 + n16] = (bf16_t)fmaxf(c2[r], 0.0f);
    }
  }

  // ---- w12 = w11@W12 + b12 (kept f32 for scale-accumulate precision) ----
#pragma unroll
  for (int tile = 0; tile < 4; ++tile) {
    int n16 = tile * 16 + m;
    float bias = b12[n16];
    v8f c;
#pragma unroll
    for (int r = 0; r < 8; ++r) c[r] = bias;
#pragma unroll
    for (int ks = 0; ks < 2; ++ks) {
      v16bf a  = load_a_frag(sW11o, 64, ks * 32, m, kh);
      v16bf bb = load_b_frag(sW12T, 64, n16, ks * 32, kh);
      c = wmma_bf16(a, bb, c);
    }
#pragma unroll
    for (int r = 0; r < 8; ++r) sw12[(r + 8 * kh) * 64 + n16] = c[r];
  }

  // A fragments of w21: reused across all 64 i-steps, keep in registers.
  v16bf a21_0 = load_a_frag(sW21o, 64, 0,  m, kh);
  v16bf a21_1 = load_a_frag(sW21o, 64, 32, m, kh);

  wait_async0();    // stage 0 landed in LDS
  __syncthreads();  // visible to all waves

  // ---- main loop: x[t,o] = sum_i w12[t,i] * (w21@W22_i + b22_i) ----
  v8f xacc[4];
#pragma unroll
  for (int t4 = 0; t4 < 4; ++t4)
#pragma unroll
    for (int r = 0; r < 8; ++r) xacc[t4][r] = 0.0f;

  for (int i = 0; i < 64; ++i) {
    const bf16_t* buf = sStage + (size_t)(i & 1) * 4096;
    if (i + 1 < 64) {  // async-prefetch next 8KB tile into the other buffer
      const char* src = (const char*)(W22T + (size_t)(i + 1) * 4096) +
                        (size_t)tid * 16;
      unsigned int dst = stage_off0 + (unsigned int)((i + 1) & 1) * 8192u;
      async_g2l_b128(dst,         src);
      async_g2l_b128(dst + 4096u, src + 4096);
    }
    float sc[8];
#pragma unroll
    for (int r = 0; r < 8; ++r) sc[r] = sw12[(r + 8 * kh) * 64 + i];
#pragma unroll
    for (int tile = 0; tile < 4; ++tile) {
      int n16 = tile * 16 + m;
      float bias = b22[i * 64 + n16];
      v8f t8;
#pragma unroll
      for (int r = 0; r < 8; ++r) t8[r] = bias;
      v16bf bb0 = load_b_frag(buf, 64, n16, 0, kh);
      t8 = wmma_bf16(a21_0, bb0, t8);
      v16bf bb1 = load_b_frag(buf, 64, n16, 32, kh);
      t8 = wmma_bf16(a21_1, bb1, t8);
#pragma unroll
      for (int r = 0; r < 8; ++r) xacc[tile][r] = fmaf(sc[r], t8[r], xacc[tile][r]);
    }
    wait_async0();    // next stage landed (barrier does not cover ASYNCcnt)
    __syncthreads();
  }

  // ---- LayerNorm over the 64-dim (N striped across 16 lanes x 4 tiles) ----
  float mu[8], rs[8];
  {
    v8f s = xacc[0] + xacc[1] + xacc[2] + xacc[3];
    v8f q = xacc[0] * xacc[0] + xacc[1] * xacc[1] +
            xacc[2] * xacc[2] + xacc[3] * xacc[3];
#pragma unroll
    for (int r = 0; r < 8; ++r) {
      float ss = s[r], qq = q[r];
#pragma unroll
      for (int msk = 1; msk <= 8; msk <<= 1) {  // stays inside 16-lane half
        ss += __shfl_xor(ss, msk, 32);
        qq += __shfl_xor(qq, msk, 32);
      }
      float mean = ss * (1.0f / 64.0f);
      float var  = qq * (1.0f / 64.0f) - mean * mean;
      mu[r] = mean;
      rs[r] = rsqrtf(var + 1e-5f);
    }
  }
#pragma unroll
  for (int tile = 0; tile < 4; ++tile) {
    int n16 = tile * 16 + m;
    float g = lng[n16], bta = lnb[n16];
#pragma unroll
    for (int r = 0; r < 8; ++r) {
      float y = (xacc[tile][r] - mu[r]) * rs[r] * g + bta;
      sXr[(r + 8 * kh) * 64 + n16] = (bf16_t)fmaxf(y, 0.0f);
    }
  }

  // ---- final: out = relu_x @ W3 + b3 ----
#pragma unroll
  for (int tile = 0; tile < 4; ++tile) {
    int n16 = tile * 16 + m;
    float bias = b3[n16];
    v8f c;
#pragma unroll
    for (int r = 0; r < 8; ++r) c[r] = bias;
#pragma unroll
    for (int ks = 0; ks < 2; ++ks) {
      v16bf a  = load_a_frag(sXr, 64, ks * 32, m, kh);
      v16bf bb = load_b_frag(sW3T, 64, n16, ks * 32, kh);
      c = wmma_bf16(a, bb, c);
    }
#pragma unroll
    for (int r = 0; r < 8; ++r)
      out[(size_t)(tok0 + r + 8 * kh) * 64 + n16] = c[r];
  }
}

extern "C" void kernel_launch(void* const* d_in, const int* in_sizes, int n_in,
                              void* d_out, int out_size, void* d_ws, size_t ws_size,
                              hipStream_t stream) {
  const float* img = (const float*)d_in[0];
  const float* loc = (const float*)d_in[1];
  const float* W11 = (const float*)d_in[2];
  const float* b11 = (const float*)d_in[3];
  const float* W12 = (const float*)d_in[4];
  const float* b12 = (const float*)d_in[5];
  const float* W21 = (const float*)d_in[6];
  const float* b21 = (const float*)d_in[7];
  const float* W22 = (const float*)d_in[8];
  const float* b22 = (const float*)d_in[9];
  const float* lng = (const float*)d_in[10];
  const float* lnb = (const float*)d_in[11];
  const float* W3  = (const float*)d_in[12];
  const float* b3  = (const float*)d_in[13];
  float* out = (float*)d_out;
  bf16_t* W22T = (bf16_t*)d_ws;  // 4096*64 bf16 = 512 KB, L2-resident

  (void)hipFuncSetAttribute((const void*)dyn_mlp_fused_kernel,
                            hipFuncAttributeMaxDynamicSharedMemorySize, SMEM_BYTES);

  // one-time weight transpose+downconvert (same stream, ordered)
  w22_transpose_kernel<<<(4096 * 64) / 256, 256, 0, stream>>>(W22, W22T);

  int n_tok = in_sizes[0] / 64;       // B*N = 32768
  int grid  = n_tok / TOK_PER_WG;     // 256 workgroups
  dyn_mlp_fused_kernel<<<grid, 256, SMEM_BYTES, stream>>>(
      img, loc, W11, b11, W12, b12, W21, b21, b22, lng, lnb, W3, b3, W22T, out);
}
